// ClusterMemory_47304769798948
// MI455X (gfx1250) — compile-verified, matrix operand order checked by reading, not run on hardware
//
#include <hip/hip_runtime.h>
#include <hip/hip_bf16.h>
#include <math.h>

// ClusterMemory CE loss: logits = inputs @ features^T / 0.05, loss = mean NLL.
// B=64, N=65536, D=2048. Bandwidth-bound (512MB of features @ 23.3TB/s ~= 22us).
// Math via v_wmma_f32_16x16x32_bf16 with hi/lo split (3 WMMAs) for ~fp32 accuracy.
// A (inputs) is pre-converted/pre-swizzled into WMMA lane layout so the hot loop
// only converts the streamed B (features) tiles.

typedef __attribute__((ext_vector_type(16))) __bf16 v16bf;
typedef __attribute__((ext_vector_type(8)))  float  v8f;
typedef __attribute__((ext_vector_type(4)))  float  f32x4;

#define BB 64
#define NN 65536
#define DD 2048
#define INV_TEMP 20.0f
#define COLS_PER_WAVE 32                  /* 2 N-tiles of 16 */
#define COLS_PER_BLOCK 256                /* 8 waves * 32 cols */
#define NBLOCKS (NN / COLS_PER_BLOCK)     /* 256 */
#define KSTEPS (DD / 32)                  /* 64 */
#define LDS_STRIDE 260

// ---- A prepack: inputs[64][2048] f32 -> per-lane WMMA operand order, hi/lo bf16.
// apack element layout: [(k0i*4 + mt)*32 + lane] * 32 bf16 : {hi[0..15], lo[0..15]}
// where v16bf element e maps to K = k0i*32 + (lane>>4)*8 + (e<8 ? e : 8+e),
// row = mt*16 + (lane&15)  (16-bit A-matrix 16x32 layout, ISA 7.12.2).
__global__ __launch_bounds__(256) void cm_pack_a_kernel(
    const float* __restrict__ inputs, __bf16* __restrict__ apack) {
  const int tid  = blockIdx.x * 256 + threadIdx.x;  // 0 .. 131071
  const int e    = tid & 15;
  const int lane = (tid >> 4) & 31;
  const int mt   = (tid >> 9) & 3;
  const int k0i  = tid >> 11;                       // 0..63
  const int row  = mt * 16 + (lane & 15);
  const int kh   = (lane >> 4) << 3;
  const int col  = k0i * 32 + kh + (e < 8 ? e : 8 + e);
  const float x  = inputs[(size_t)row * DD + col];
  const __bf16 h = (__bf16)x;
  const __bf16 l = (__bf16)(x - (float)h);
  const size_t base = ((size_t)(k0i * 4 + mt) * 32 + lane) * 32;
  apack[base + e]      = h;
  apack[base + 16 + e] = l;
}

// Load one 16x32 f32 B-tile slice for this lane and split to hi/lo bf16.
__device__ __forceinline__ void load_cvt(const float* __restrict__ base,
                                         v16bf& hi, v16bf& lo) {
  f32x4 q0 = *(const f32x4*)(base + 0);
  f32x4 q1 = *(const f32x4*)(base + 4);
  f32x4 q2 = *(const f32x4*)(base + 16);
  f32x4 q3 = *(const f32x4*)(base + 20);
  float f[16];
#pragma unroll
  for (int i = 0; i < 4; ++i) {
    f[i]      = q0[i];
    f[4 + i]  = q1[i];
    f[8 + i]  = q2[i];
    f[12 + i] = q3[i];
  }
#pragma unroll
  for (int i = 0; i < 16; ++i) {
    float x  = f[i];
    __bf16 h = (__bf16)x;            // RNE
    hi[i]    = h;
    lo[i]    = (__bf16)(x - (float)h);
  }
}

__global__ __launch_bounds__(256) void cm_gemm_partials_kernel(
    const __bf16* __restrict__ apack,    // prepacked inputs (hi/lo)
    const int*    __restrict__ targets,  // [64]
    const float*  __restrict__ features, // [65536, 2048]
    float2*       __restrict__ partials, // [NBLOCKS, 64] (rowmax, sumexp)
    float*        __restrict__ tlogit) { // [64] logit at target column
  __shared__ float lds[64 * LDS_STRIDE];
  __shared__ float pm[256];
  __shared__ float ps[256];

  const int lane = threadIdx.x & 31;
  const int wave = threadIdx.x >> 5;
  const int r    = lane & 15;
  const int kh   = (lane >> 4) << 3;
  const int n0   = blockIdx.x * COLS_PER_BLOCK + wave * COLS_PER_WAVE;

  const float* Fb0 = features + (size_t)(n0 + r) * DD + kh;
  const float* Fb1 = Fb0 + (size_t)16 * DD;
  const __bf16* Ab = apack + (size_t)lane * 32;

  v8f acc[4][2];
#pragma unroll
  for (int mt = 0; mt < 4; ++mt) {
    acc[mt][0] = (v8f)0.0f;
    acc[mt][1] = (v8f)0.0f;
  }

  for (int k0i = 0; k0i < KSTEPS; ++k0i) {
    const int k0 = k0i * 32;
    v16bf bhi0, blo0, bhi1, blo1;
    load_cvt(Fb0 + k0, bhi0, blo0);
    load_cvt(Fb1 + k0, bhi1, blo1);
    __builtin_prefetch(Fb0 + k0 + 128, 0, 1);  // global_prefetch_b8
    __builtin_prefetch(Fb1 + k0 + 128, 0, 1);
#pragma unroll
    for (int mt = 0; mt < 4; ++mt) {
      const __bf16* ap = Ab + (size_t)(k0i * 4 + mt) * 1024;
      v16bf ahi = *(const v16bf*)(ap);        // 2x b128 loads
      v16bf alo = *(const v16bf*)(ap + 16);
      // 3-term hi/lo split (lo*lo dropped: O(eps^2)) for both N-tiles
      acc[mt][0] = __builtin_amdgcn_wmma_f32_16x16x32_bf16(
          false, ahi, false, bhi0, (short)0, acc[mt][0], false, false);
      acc[mt][0] = __builtin_amdgcn_wmma_f32_16x16x32_bf16(
          false, ahi, false, blo0, (short)0, acc[mt][0], false, false);
      acc[mt][0] = __builtin_amdgcn_wmma_f32_16x16x32_bf16(
          false, alo, false, bhi0, (short)0, acc[mt][0], false, false);
      acc[mt][1] = __builtin_amdgcn_wmma_f32_16x16x32_bf16(
          false, ahi, false, bhi1, (short)0, acc[mt][1], false, false);
      acc[mt][1] = __builtin_amdgcn_wmma_f32_16x16x32_bf16(
          false, ahi, false, blo1, (short)0, acc[mt][1], false, false);
      acc[mt][1] = __builtin_amdgcn_wmma_f32_16x16x32_bf16(
          false, alo, false, bhi1, (short)0, acc[mt][1], false, false);
    }
  }

  // Scatter scaled logits to LDS: lanes 0-15 hold rows v, lanes 16-31 rows v+8.
  const int cl = wave * COLS_PER_WAVE + (lane & 15);
  const int rv = (lane >> 4) << 3;
#pragma unroll
  for (int mt = 0; mt < 4; ++mt) {
#pragma unroll
    for (int v = 0; v < 8; ++v) {
      lds[(mt * 16 + rv + v) * LDS_STRIDE + cl]      = acc[mt][0][v] * INV_TEMP;
      lds[(mt * 16 + rv + v) * LDS_STRIDE + cl + 16] = acc[mt][1][v] * INV_TEMP;
    }
  }
  __syncthreads();

  // Phase 1: 256 threads, each reduces 64 of this block's 256 columns for one row.
  {
    const int row = threadIdx.x & 63;
    const int q   = threadIdx.x >> 6;       // 0..3
    const float* rp = &lds[row * LDS_STRIDE + q * 64];
    float m = -INFINITY;
    for (int c = 0; c < 64; ++c) m = fmaxf(m, rp[c]);
    float s = 0.0f;
    for (int c = 0; c < 64; ++c) s += __expf(rp[c] - m);
    pm[q * 64 + row] = m;
    ps[q * 64 + row] = s;
  }
  __syncthreads();

  // Phase 2: 64 threads merge the 4 quarter-partials, emit block partial + target.
  if (threadIdx.x < 64) {
    const int row = threadIdx.x;
    float M = -INFINITY, S = 0.0f;
#pragma unroll
    for (int q = 0; q < 4; ++q) {
      float m  = pm[q * 64 + row];
      float s  = ps[q * 64 + row];
      float nm = fmaxf(M, m);
      S = S * __expf(M - nm) + s * __expf(m - nm);
      M = nm;
    }
    partials[(size_t)blockIdx.x * 64 + row] = make_float2(M, S);

    const int t  = targets[row];
    const int nb = blockIdx.x * COLS_PER_BLOCK;
    if (t >= nb && t < nb + COLS_PER_BLOCK)
      tlogit[row] = lds[row * LDS_STRIDE + (t - nb)];
  }
}

__global__ __launch_bounds__(64) void cm_loss_kernel(
    const float2* __restrict__ partials,
    const float*  __restrict__ tlogit,
    float*        __restrict__ out) {
  __shared__ float red[64];
  const int b = threadIdx.x;
  float M = -INFINITY, S = 0.0f;
  for (int p = 0; p < NBLOCKS; ++p) {
    float2 ms = partials[(size_t)p * 64 + b];
    float nm  = fmaxf(M, ms.x);
    S = S * __expf(M - nm) + ms.y * __expf(ms.x - nm);
    M = nm;
  }
  red[b] = -(tlogit[b] - (M + __logf(S)));
  __syncthreads();
  for (int off = 32; off > 0; off >>= 1) {
    if (b < off) red[b] += red[b + off];
    __syncthreads();
  }
  if (b == 0) out[0] = red[0] * (1.0f / (float)BB);
}

extern "C" void kernel_launch(void* const* d_in, const int* in_sizes, int n_in,
                              void* d_out, int out_size, void* d_ws, size_t ws_size,
                              hipStream_t stream) {
  const float* inputs   = (const float*)d_in[0];
  const int*   targets  = (const int*)d_in[1];
  const float* features = (const float*)d_in[2];

  // ws layout: apack (64*2048*2 bf16 = 512 KiB) | partials (256*64 float2) | tlogit (64 f32)
  char* ws = (char*)d_ws;
  __bf16* apack    = (__bf16*)ws;
  float2* partials = (float2*)(ws + (size_t)BB * DD * 2 * sizeof(__bf16));
  float*  tlogit   = (float*)((char*)partials + (size_t)NBLOCKS * 64 * sizeof(float2));

  cm_pack_a_kernel<<<512, 256, 0, stream>>>(inputs, apack);
  cm_gemm_partials_kernel<<<NBLOCKS, 256, 0, stream>>>(
      apack, targets, features, partials, tlogit);
  cm_loss_kernel<<<1, 64, 0, stream>>>(partials, tlogit, (float*)d_out);
}
